// EUCFConv_44590350467104
// MI455X (gfx1250) — matrix operand order repaired and checked by reading, not compiled, not run on hardware
//
#include <hip/hip_runtime.h>
#include <cstddef>

#define DIM      64
#define RBF      64
#define N_NODES  100000
#define N_EDGES  1600000

typedef __attribute__((ext_vector_type(16))) _Float16 v16h;
typedef __attribute__((ext_vector_type(8)))  _Float16 v8h;
typedef __attribute__((ext_vector_type(4)))  _Float16 v4h;
typedef __attribute__((ext_vector_type(8)))  float    v8f;

// shifted softplus: log(1+exp(x)) - log(2)
static __device__ __forceinline__ float ssp_act(float x) {
    float sp = (x > 15.0f) ? x : __logf(1.0f + __expf(x));
    return sp - 0.6931471805599453f;
}

// Gather a 16x32 f16 A-fragment from a row-major [16 x 64] f16 LDS tile.
// CDNA5 layout: lanes 0-15 (M=lane): VGPR0-3 = K 0..7, VGPR4-7 = K 16..23
//               lanes 16-31 (M=lane-16): VGPR0-3 = K 8..15, VGPR4-7 = K 24..31
// kt selects K-chunk 0 (K 0..31) or 1 (K 32..63).
static __device__ __forceinline__ v16h load_afrag(const _Float16* s, int lane, int kt) {
    int m  = lane & 15;
    int kb = (lane < 16) ? 0 : 8;
    int k0 = kt * 32 + kb;
    v8h lo = *(const v8h*)(s + m * 64 + k0);
    v8h hi = *(const v8h*)(s + m * 64 + k0 + 16);
    v16h r;
#pragma unroll
    for (int i = 0; i < 8; ++i) { r[i] = lo[i]; r[i + 8] = hi[i]; }
    return r;
}

// B-fragment (32x16, K-major pairs per VGPR) from a TRANSPOSED weight tile
// wt[n*64 + k] (f16). Lanes 0-15: K 0..7,16..23, lanes 16-31: K 8..15,24..31 of the chunk,
// matching the A-fragment layout -> two 16-byte LDS loads.
static __device__ __forceinline__ v16h load_bfrag(const _Float16* wt, int lane, int nt, int kt) {
    int n  = nt * 16 + (lane & 15);
    int kb = (lane < 16) ? 0 : 8;
    v8h lo = *(const v8h*)(wt + n * 64 + kt * 32 + kb);
    v8h hi = *(const v8h*)(wt + n * 64 + kt * 32 + kb + 16);
    v16h r;
#pragma unroll
    for (int i = 0; i < 8; ++i) { r[i] = lo[i]; r[i + 8] = hi[i]; }
    return r;
}

// Stage W[k][n] (row-major f32 64x64) into LDS transposed as f16: wt[n][k].
static __device__ __forceinline__ void stage_weight_t(_Float16* wt, const float* W, int tid) {
    for (int i = tid; i < 64 * 64; i += 256) {
        int n = i & 63, k = i >> 6;          // coalesced global reads over n
        wt[n * 64 + k] = (_Float16)W[k * 64 + n];
    }
}

__global__ void zero_kernel(float4* __restrict__ p, int n4) {
    int i = blockIdx.x * 256 + threadIdx.x;
    if (i < n4) p[i] = make_float4(0.f, 0.f, 0.f, 0.f);
}

// Fused edge MLP + scatter-sum:  h[dst] += ssp(E @ We1 + b1) @ We2 + b2
__global__ __launch_bounds__(256) void schnet_edge_kernel(
    const float* __restrict__ edge_feats,
    const int*   __restrict__ dst,
    const float* __restrict__ W_e1, const float* __restrict__ b_e1,
    const float* __restrict__ W_e2, const float* __restrict__ b_e2,
    float* __restrict__ h)
{
    __shared__ __attribute__((aligned(32))) _Float16 wt1[64 * 64];
    __shared__ __attribute__((aligned(32))) _Float16 wt2[64 * 64];
    __shared__ __attribute__((aligned(32))) _Float16 stage[8][16 * 64];

    const int tid = threadIdx.x;
    stage_weight_t(wt1, W_e1, tid);
    stage_weight_t(wt2, W_e2, tid);
    __syncthreads();

    const int wave  = tid >> 5;
    const int lane  = tid & 31;
    const int n     = lane & 15;
    const int mbase = (lane < 16) ? 0 : 8;
    _Float16* st = stage[wave];

    float bias1[4], bias2[4];
#pragma unroll
    for (int t = 0; t < 4; ++t) { bias1[t] = b_e1[t * 16 + n]; bias2[t] = b_e2[t * 16 + n]; }

    const int n_tiles = N_EDGES / 16;
    const int stride  = gridDim.x * 8;
    for (int tile = blockIdx.x * 8 + wave; tile < n_tiles; tile += stride) {
        // ---- stage 16x64 f32 edge tile -> f16 LDS (coalesced b128 loads) ----
        const float4* src = (const float4*)(edge_feats + (size_t)tile * (16 * 64));
#pragma unroll
        for (int it = 0; it < 8; ++it) {
            int idx  = it * 32 + lane;
            float4 f = src[idx];
            v4h hv; hv[0] = (_Float16)f.x; hv[1] = (_Float16)f.y;
                    hv[2] = (_Float16)f.z; hv[3] = (_Float16)f.w;
            *(v4h*)(st + idx * 4) = hv;
        }
        asm volatile("s_wait_dscnt 0" ::: "memory");

        v16h a0 = load_afrag(st, lane, 0);
        v16h a1 = load_afrag(st, lane, 1);
        int my_dst = dst[tile * 16 + n];
        asm volatile("s_wait_dscnt 0" ::: "memory");

        // ---- GEMM1 (K=64 via two 16x16x32 WMMA) + ssp -> LDS transpose ----
#pragma unroll
        for (int nt = 0; nt < 4; ++nt) {
            v8f acc;
#pragma unroll
            for (int r = 0; r < 8; ++r) acc[r] = bias1[nt];
            v16h b0 = load_bfrag(wt1, lane, nt, 0);
            v16h b1 = load_bfrag(wt1, lane, nt, 1);
            acc = __builtin_amdgcn_wmma_f32_16x16x32_f16(false, a0, false, b0, (short)0, acc, false, false);
            acc = __builtin_amdgcn_wmma_f32_16x16x32_f16(false, a1, false, b1, (short)0, acc, false, false);
#pragma unroll
            for (int r = 0; r < 8; ++r)
                st[(mbase + r) * 64 + nt * 16 + n] = (_Float16)ssp_act(acc[r]);
        }
        asm volatile("s_wait_dscnt 0" ::: "memory");

        v16h t0 = load_afrag(st, lane, 0);
        v16h t1 = load_afrag(st, lane, 1);

        // ---- GEMM2 + bias -> atomic scatter to h[dst] ----
#pragma unroll
        for (int nt = 0; nt < 4; ++nt) {
            v8f acc;
#pragma unroll
            for (int r = 0; r < 8; ++r) acc[r] = bias2[nt];
            v16h b0 = load_bfrag(wt2, lane, nt, 0);
            v16h b1 = load_bfrag(wt2, lane, nt, 1);
            acc = __builtin_amdgcn_wmma_f32_16x16x32_f16(false, t0, false, b0, (short)0, acc, false, false);
            acc = __builtin_amdgcn_wmma_f32_16x16x32_f16(false, t1, false, b1, (short)0, acc, false, false);
#pragma unroll
            for (int r = 0; r < 8; ++r) {
                int d = __shfl(my_dst, mbase + r, 32);
                unsafeAtomicAdd(h + (size_t)d * 64 + nt * 16 + n, acc[r]);
            }
        }
    }
}

// Node MLP:  out = ssp(h @ Wo1 + b1) @ Wo2 + b2
__global__ __launch_bounds__(256) void schnet_node_kernel(
    const float* __restrict__ hin,
    const float* __restrict__ W_o1, const float* __restrict__ b_o1,
    const float* __restrict__ W_o2, const float* __restrict__ b_o2,
    float* __restrict__ out)
{
    __shared__ __attribute__((aligned(32))) _Float16 wt1[64 * 64];
    __shared__ __attribute__((aligned(32))) _Float16 wt2[64 * 64];
    __shared__ __attribute__((aligned(32))) _Float16 stage[8][16 * 64];

    const int tid = threadIdx.x;
    stage_weight_t(wt1, W_o1, tid);
    stage_weight_t(wt2, W_o2, tid);
    __syncthreads();

    const int wave  = tid >> 5;
    const int lane  = tid & 31;
    const int n     = lane & 15;
    const int mbase = (lane < 16) ? 0 : 8;
    _Float16* st = stage[wave];

    float bias1[4], bias2[4];
#pragma unroll
    for (int t = 0; t < 4; ++t) { bias1[t] = b_o1[t * 16 + n]; bias2[t] = b_o2[t * 16 + n]; }

    const int n_tiles = N_NODES / 16;     // 6250
    const int stride  = gridDim.x * 8;
    for (int tile = blockIdx.x * 8 + wave; tile < n_tiles; tile += stride) {
        const float4* src = (const float4*)(hin + (size_t)tile * (16 * 64));
#pragma unroll
        for (int it = 0; it < 8; ++it) {
            int idx  = it * 32 + lane;
            float4 f = src[idx];
            v4h hv; hv[0] = (_Float16)f.x; hv[1] = (_Float16)f.y;
                    hv[2] = (_Float16)f.z; hv[3] = (_Float16)f.w;
            *(v4h*)(st + idx * 4) = hv;
        }
        asm volatile("s_wait_dscnt 0" ::: "memory");

        v16h a0 = load_afrag(st, lane, 0);
        v16h a1 = load_afrag(st, lane, 1);
        asm volatile("s_wait_dscnt 0" ::: "memory");

#pragma unroll
        for (int nt = 0; nt < 4; ++nt) {
            v8f acc;
#pragma unroll
            for (int r = 0; r < 8; ++r) acc[r] = bias1[nt];
            v16h b0 = load_bfrag(wt1, lane, nt, 0);
            v16h b1 = load_bfrag(wt1, lane, nt, 1);
            acc = __builtin_amdgcn_wmma_f32_16x16x32_f16(false, a0, false, b0, (short)0, acc, false, false);
            acc = __builtin_amdgcn_wmma_f32_16x16x32_f16(false, a1, false, b1, (short)0, acc, false, false);
#pragma unroll
            for (int r = 0; r < 8; ++r)
                st[(mbase + r) * 64 + nt * 16 + n] = (_Float16)ssp_act(acc[r]);
        }
        asm volatile("s_wait_dscnt 0" ::: "memory");

        v16h t0 = load_afrag(st, lane, 0);
        v16h t1 = load_afrag(st, lane, 1);

#pragma unroll
        for (int nt = 0; nt < 4; ++nt) {
            v8f acc;
#pragma unroll
            for (int r = 0; r < 8; ++r) acc[r] = bias2[nt];
            v16h b0 = load_bfrag(wt2, lane, nt, 0);
            v16h b1 = load_bfrag(wt2, lane, nt, 1);
            acc = __builtin_amdgcn_wmma_f32_16x16x32_f16(false, t0, false, b0, (short)0, acc, false, false);
            acc = __builtin_amdgcn_wmma_f32_16x16x32_f16(false, t1, false, b1, (short)0, acc, false, false);
#pragma unroll
            for (int r = 0; r < 8; ++r)
                out[(size_t)(tile * 16 + mbase + r) * 64 + nt * 16 + n] = acc[r];
        }
    }
}

extern "C" void kernel_launch(void* const* d_in, const int* in_sizes, int n_in,
                              void* d_out, int out_size, void* d_ws, size_t ws_size,
                              hipStream_t stream)
{
    (void)in_sizes; (void)n_in; (void)out_size; (void)ws_size;
    // setup_inputs order:
    // 0 node_feats, 1 edge_feats, 2 dst, 3 W_n1, 4 b_n1,
    // 5 W_e1, 6 b_e1, 7 W_e2, 8 b_e2, 9 W_o1, 10 b_o1, 11 W_o2, 12 b_o2
    const float* edge_feats = (const float*)d_in[1];
    const int*   dst        = (const int*)d_in[2];
    const float* W_e1 = (const float*)d_in[5];
    const float* b_e1 = (const float*)d_in[6];
    const float* W_e2 = (const float*)d_in[7];
    const float* b_e2 = (const float*)d_in[8];
    const float* W_o1 = (const float*)d_in[9];
    const float* b_o1 = (const float*)d_in[10];
    const float* W_o2 = (const float*)d_in[11];
    const float* b_o2 = (const float*)d_in[12];

    float* h   = (float*)d_ws;    // [N_NODES, DIM] f32 accumulator (25.6 MB)
    float* out = (float*)d_out;

    const int n4 = N_NODES * DIM / 4;
    zero_kernel<<<(n4 + 255) / 256, 256, 0, stream>>>((float4*)h, n4);

    schnet_edge_kernel<<<625, 256, 0, stream>>>(edge_feats, dst, W_e1, b_e1, W_e2, b_e2, h);
    schnet_node_kernel<<<256, 256, 0, stream>>>(h, W_o1, b_o1, W_o2, b_o2, out);
}
